// TransformerDecoderLayer_8117488190255
// MI455X (gfx1250) — matrix operand order repaired and checked by reading, not compiled, not run on hardware
//
#include <hip/hip_runtime.h>
#include <hip/hip_bf16.h>

// ---------------------------------------------------------------------------
// CDNA5 (gfx1250) deformable-DETR decoder layer, bf16 WMMA everywhere.
// - Weights pre-transposed to [N,K] bf16: every B-frag = two 16B global loads
// - bf16 activation pipeline: intermediates stored bf16 (halves HBM traffic);
//   f32 kept for accumulators, residuals, LN stats, softmax, sampling coords
// - A tiles double-buffered through LDS (one barrier/K-step) + prefetch
// - Residual+LayerNorm fused into GEMM epilogue (DPP row sums + tiny LDS)
// - Flash attention: 1/sqrt(HD) folded into Wq/bq; DPP softmax reductions
// ---------------------------------------------------------------------------

typedef __bf16 bf16_t;
typedef __attribute__((ext_vector_type(16))) __bf16 v16bf;
typedef __attribute__((ext_vector_type(8)))  __bf16 v8bf;
typedef __attribute__((ext_vector_type(4)))  __bf16 v4bf;
typedef __attribute__((ext_vector_type(8)))  float  v8f;

#define D_    256
#define NH_   8
#define HD_   32
#define DFF_  1024
#define B_    32
#define Q_    300
#define NQ_   (B_ * Q_)       // 9600
#define LMEM_ 8500
#define NM_   (B_ * LMEM_)    // 272000

__device__ __forceinline__ v16bf ld16_split(const bf16_t* p0, const bf16_t* p1) {
  v8bf lo = *(const v8bf*)p0;
  v8bf hi = *(const v8bf*)p1;
  return __builtin_shufflevector(lo, hi, 0, 1, 2, 3, 4, 5, 6, 7,
                                 8, 9, 10, 11, 12, 13, 14, 15);
}

// ---- DPP 16-lane reductions (VALU only, no LDS round-trips) ----
template <int CTRL>
__device__ __forceinline__ float dpp_xf(float v) {
  int x = __float_as_int(v);
  int y = __builtin_amdgcn_update_dpp(x, x, CTRL, 0xF, 0xF, false);
  return __int_as_float(y);
}
// quad_perm(1,0,3,2)=0xB1, quad_perm(2,3,0,1)=0x4E, row_ror:4=0x124, row_ror:8=0x128
__device__ __forceinline__ float red16_max(float v) {
  v = fmaxf(v, dpp_xf<0xB1>(v));
  v = fmaxf(v, dpp_xf<0x4E>(v));
  v = fmaxf(v, dpp_xf<0x124>(v));
  v = fmaxf(v, dpp_xf<0x128>(v));
  return v;
}
__device__ __forceinline__ float red16_sum(float v) {
  v = v + dpp_xf<0xB1>(v);
  v = v + dpp_xf<0x4E>(v);
  v = v + dpp_xf<0x124>(v);
  v = v + dpp_xf<0x128>(v);
  return v;
}

// ------------------------------ small kernels ------------------------------

// f32 [K,N] -> bf16 [N,K] (transpose + convert + optional scale)
__global__ void w_to_bf16_t(const float* __restrict__ W, bf16_t* __restrict__ Wt,
                            int K, int N, float scale) {
  int i = blockIdx.x * blockDim.x + threadIdx.x;
  if (i >= N * K) return;
  int n = i / K, k = i - n * K;
  Wt[i] = (bf16_t)(W[(size_t)k * N + n] * scale);
}

__global__ void scale_kernel(const float* __restrict__ s, float* __restrict__ d,
                             float sc, int n) {
  int i = blockIdx.x * blockDim.x + threadIdx.x;
  if (i < n) d[i] = s[i] * sc;
}

__global__ void add_bf16_kernel(const float* __restrict__ a,
                                const float* __restrict__ b,
                                bf16_t* __restrict__ o, int n) {
  int i = blockIdx.x * blockDim.x + threadIdx.x;
  if (i < n) o[i] = (bf16_t)(a[i] + b[i]);
}

__global__ void softmax16_kernel(float* __restrict__ p, int n) {
  int i = blockIdx.x * blockDim.x + threadIdx.x;
  if (i >= n) return;
  float4* row = (float4*)(p + (size_t)i * 16);
  float4 v[4];
  v[0] = row[0]; v[1] = row[1]; v[2] = row[2]; v[3] = row[3];
  float m = -1e30f;
#pragma unroll
  for (int j = 0; j < 4; ++j)
    m = fmaxf(m, fmaxf(fmaxf(v[j].x, v[j].y), fmaxf(v[j].z, v[j].w)));
  float s = 0.f;
#pragma unroll
  for (int j = 0; j < 4; ++j) {
    v[j].x = __expf(v[j].x - m); v[j].y = __expf(v[j].y - m);
    v[j].z = __expf(v[j].z - m); v[j].w = __expf(v[j].w - m);
    s += v[j].x + v[j].y + v[j].z + v[j].w;
  }
  float inv = 1.f / s;
#pragma unroll
  for (int j = 0; j < 4; ++j) {
    v[j].x *= inv; v[j].y *= inv; v[j].z *= inv; v[j].w *= inv;
    row[j] = v[j];
  }
}

// ------------------------------ WMMA GEMM ----------------------------------
// C[M,N] = A[M,K] * W[K,N], W transposed as Wt[N,K] (bf16).
// ABF: A is bf16 (pure b128 copy staging); else A is f32 (convert in staging).
// Block: 64 rows x CT*32 cols, 8 waves (4 row-tiles x 2 col-halves),
// CT accumulators/wave, K-step 32, double-buffered A tile, B from global.
// LNF: fused  out = LayerNorm(res + C + bias) * gamma + beta  (N==256,
// grid.x==1, CT==8 so a block owns complete rows).
template <int CT, bool RELU, bool OUTB, bool LNF, bool ABF>
__global__ __launch_bounds__(256) void gemm_tn(
    const float* __restrict__ Af, const bf16_t* __restrict__ Ab,
    const bf16_t* __restrict__ Wt, const float* __restrict__ bias,
    float* __restrict__ outF, bf16_t* __restrict__ outB,
    int M, int N, int K,
    const float* __restrict__ res, const float* __restrict__ gamma,
    const float* __restrict__ lbeta) {
  __shared__ alignas(16) bf16_t As[2][64][40];
  const int tid  = threadIdx.x;
  const int lane = tid & 31;
  const int wv   = tid >> 5;
  const int m0   = blockIdx.y * 64;
  const int n0   = blockIdx.x * (CT * 32);
  const int mt   = wv & 3;
  const int nh   = wv >> 2;
  const int colbase = n0 + nh * (CT * 16);
  const int row  = lane & 15;
  const int koA  = (lane >= 16) ? 8 : 0;   // A-frag half-K offset
  const int col  = lane & 15;
  const int koB  = (lane >= 16) ? 8 : 0;   // B-frag half-K offset
  v8f acc[CT];
#pragma unroll
  for (int ct = 0; ct < CT; ++ct) acc[ct] = {};

  auto stageA = [&](int buf, int kk) {
    if constexpr (ABF) {                   // 2048 bf16: one v8bf per thread
      int r = tid >> 2, c8 = (tid & 3) * 8;
      *(v8bf*)(&As[buf][r][c8]) =
          *(const v8bf*)(Ab + (size_t)(m0 + r) * K + kk + c8);
    } else {                               // 512 float4: f32 -> bf16
#pragma unroll
      for (int i = 0; i < 2; ++i) {
        int e4 = tid + i * 256;
        int r = e4 >> 3, c4 = e4 & 7;
        float4 v4 = *(const float4*)(Af + (size_t)(m0 + r) * K + kk + c4 * 4);
        v4bf s;
        s[0] = (bf16_t)v4.x; s[1] = (bf16_t)v4.y;
        s[2] = (bf16_t)v4.z; s[3] = (bf16_t)v4.w;
        *(v4bf*)(&As[buf][r][c4 * 4]) = s;
      }
    }
  };

  stageA(0, 0);
  int p = 0;
  for (int kk = 0; kk < K; kk += 32) {
    __syncthreads();                       // buffer p ready for everyone
    if (kk + 32 < K) stageA(p ^ 1, kk + 32);
    if (kk + 64 < K) {                     // prefetch 2 steps ahead
      if constexpr (ABF)
        __builtin_prefetch(Ab + (size_t)(m0 + (tid >> 2)) * K + kk + 64 + (tid & 3) * 8, 0, 1);
      else
        __builtin_prefetch(Af + (size_t)(m0 + (tid >> 2)) * K + kk + 64 + (tid & 3) * 8, 0, 1);
    }
    v16bf a = ld16_split(&As[p][mt * 16 + row][koA],
                         &As[p][mt * 16 + row][16 + koA]);
    const bf16_t* wbase = Wt + (size_t)(colbase + col) * K + kk + koB;
#pragma unroll
    for (int ct = 0; ct < CT; ++ct) {
      const bf16_t* wp = wbase + (size_t)(ct * 16) * K;
      v16bf bfr = ld16_split(wp, wp + 16);
      acc[ct] = __builtin_amdgcn_wmma_f32_16x16x32_bf16(
          false, a, false, bfr, (short)0, acc[ct], false, false);
    }
    p ^= 1;
  }

  const int rb = (lane >= 16) ? 8 : 0;
  if constexpr (LNF) {
    __shared__ float lnsum[4][16][2][2];   // [mt][rowInTile][nh][{sum,sumsq}]
    float s1[8], s2[8];
#pragma unroll
    for (int r = 0; r < 8; ++r) { s1[r] = 0.f; s2[r] = 0.f; }
#pragma unroll
    for (int ct = 0; ct < CT; ++ct) {
#pragma unroll
      for (int r = 0; r < 8; ++r) {
        int gm = m0 + mt * 16 + rb + r;
        int gn = colbase + ct * 16 + col;
        float v = acc[ct][r] + bias[gn] + res[(size_t)gm * N + gn];
        acc[ct][r] = v;
        s1[r] += v;
        s2[r] += v * v;
      }
    }
#pragma unroll
    for (int r = 0; r < 8; ++r) {          // per-half-wave row partials (DPP)
      s1[r] = red16_sum(s1[r]);
      s2[r] = red16_sum(s2[r]);
    }
    if (lane == 0 || lane == 16) {
#pragma unroll
      for (int r = 0; r < 8; ++r) {
        lnsum[mt][rb + r][nh][0] = s1[r];
        lnsum[mt][rb + r][nh][1] = s2[r];
      }
    }
    __syncthreads();
    float mu[8], rstd[8];
#pragma unroll
    for (int r = 0; r < 8; ++r) {
      float t1 = lnsum[mt][rb + r][0][0] + lnsum[mt][rb + r][1][0];
      float t2 = lnsum[mt][rb + r][0][1] + lnsum[mt][rb + r][1][1];
      float mean = t1 * (1.f / 256.f);
      float var  = t2 * (1.f / 256.f) - mean * mean;
      mu[r] = mean;
      rstd[r] = rsqrtf(var + 1e-5f);
    }
#pragma unroll
    for (int ct = 0; ct < CT; ++ct) {
#pragma unroll
      for (int r = 0; r < 8; ++r) {
        int gm = m0 + mt * 16 + rb + r;
        int gn = colbase + ct * 16 + col;
        outF[(size_t)gm * N + gn] =
            (acc[ct][r] - mu[r]) * rstd[r] * gamma[gn] + lbeta[gn];
      }
    }
  } else {
#pragma unroll
    for (int ct = 0; ct < CT; ++ct) {
#pragma unroll
      for (int r = 0; r < 8; ++r) {
        int gm = m0 + mt * 16 + rb + r;
        int gn = colbase + ct * 16 + col;
        float v = acc[ct][r] + bias[gn];
        if (RELU) v = fmaxf(v, 0.f);
        if (OUTB) outB[(size_t)gm * N + gn] = (bf16_t)v;
        else      outF[(size_t)gm * N + gn] = v;
      }
    }
  }
}

// ------------------------- flash attention (WMMA) --------------------------
// grid (5 q-tiles, NH, B), 128 threads = 4 waves x 16 query rows each.
// Q/K/V are bf16 (Q pre-scaled by 1/sqrt(HD) via the Wq/bq conversion).
// K tile [32key][32hd] and V tile transposed [32hd][32key] staged in LDS.
__global__ __launch_bounds__(128) void attn_flash(
    const bf16_t* __restrict__ Qb, const bf16_t* __restrict__ Kb,
    const bf16_t* __restrict__ Vb, bf16_t* __restrict__ Ob) {
  __shared__ alignas(16) bf16_t Kt[32][40];
  __shared__ alignas(16) bf16_t Vt[32][40];
  __shared__ alignas(16) bf16_t Plds[4][16][40];
  const int lane = threadIdx.x & 31;
  const int wv   = threadIdx.x >> 5;
  const int b = blockIdx.z, h = blockIdx.y;
  const int q0 = blockIdx.x * 64 + wv * 16;
  const int koA = (lane >= 16) ? 8 : 0;
  const int koB = (lane >= 16) ? 8 : 0;
  // Q fragment: two direct 16B global loads (already scaled)
  v16bf qa;
  {
    int qr = q0 + (lane & 15); if (qr >= Q_) qr = Q_ - 1;
    const bf16_t* qp = Qb + ((size_t)(b * Q_ + qr)) * D_ + h * HD_;
    qa = ld16_split(qp + koA, qp + 16 + koA);
  }
  v8f c0 = {}, c1 = {};
  float mrow[8], lrow[8];
#pragma unroll
  for (int r = 0; r < 8; ++r) { mrow[r] = -1e30f; lrow[r] = 0.f; }

  for (int kt = 0; kt < 10; ++kt) {          // 10 * 32 >= 300 keys
    const int kb = kt * 32;
    // --- stage K and V tiles (shared by all 4 waves) ---
    {
      int kr = threadIdx.x >> 2;             // 0..31
      int c8 = (threadIdx.x & 3) * 8;        // 0,8,16,24
      int key = kb + kr; if (key >= Q_) key = Q_ - 1;
      size_t basei = ((size_t)(b * Q_ + key)) * D_ + h * HD_ + c8;
      *(v8bf*)(&Kt[kr][c8]) = *(const v8bf*)(Kb + basei);
      v8bf vv = *(const v8bf*)(Vb + basei);
#pragma unroll
      for (int j = 0; j < 8; ++j) Vt[c8 + j][kr] = vv[j];
    }
    __syncthreads();
    // --- scores ---
    v16bf kf0 = ld16_split(&Kt[lane & 15][koB],        &Kt[lane & 15][koB] + 16);
    v16bf kf1 = ld16_split(&Kt[16 + (lane & 15)][koB], &Kt[16 + (lane & 15)][koB] + 16);
    v8f s0 = {}, s1 = {};
    s0 = __builtin_amdgcn_wmma_f32_16x16x32_bf16(false, qa, false, kf0, (short)0, s0, false, false);
    s1 = __builtin_amdgcn_wmma_f32_16x16x32_bf16(false, qa, false, kf1, (short)0, s1, false, false);
    const bool inv0 = (kb + (lane & 15)) >= Q_;
    const bool inv1 = (kb + 16 + (lane & 15)) >= Q_;
    float p0[8], p1[8];
#pragma unroll
    for (int r = 0; r < 8; ++r) {
      float v0 = inv0 ? -1e30f : s0[r];
      float v1 = inv1 ? -1e30f : s1[r];
      float tmax = red16_max(fmaxf(v0, v1));           // DPP, no LDS
      float mnew = fmaxf(mrow[r], tmax);
      float e0 = __expf(v0 - mnew), e1 = __expf(v1 - mnew);
      float rs = red16_sum(e0 + e1);                   // DPP, no LDS
      float alpha = __expf(mrow[r] - mnew);
      lrow[r] = lrow[r] * alpha + rs;
      mrow[r] = mnew;
      c0[r] *= alpha; c1[r] *= alpha;
      p0[r] = e0; p1[r] = e1;
    }
    {
      int rb2 = (lane >= 16) ? 8 : 0;
#pragma unroll
      for (int r = 0; r < 8; ++r) {          // C-layout -> LDS [row][key]
        Plds[wv][rb2 + r][lane & 15]        = (bf16_t)p0[r];
        Plds[wv][rb2 + r][16 + (lane & 15)] = (bf16_t)p1[r];
      }
    }
    // same-wave LDS ops are in order; only this wave touches Plds[wv]
    v16bf pa = ld16_split(&Plds[wv][lane & 15][koA],
                          &Plds[wv][lane & 15][16 + koA]);
    v16bf vb0 = ld16_split(&Vt[lane & 15][koB],        &Vt[lane & 15][koB] + 16);
    v16bf vb1 = ld16_split(&Vt[16 + (lane & 15)][koB], &Vt[16 + (lane & 15)][koB] + 16);
    c0 = __builtin_amdgcn_wmma_f32_16x16x32_bf16(false, pa, false, vb0, (short)0, c0, false, false);
    c1 = __builtin_amdgcn_wmma_f32_16x16x32_bf16(false, pa, false, vb1, (short)0, c1, false, false);
    __syncthreads();
  }
  {
    int col = lane & 15;
    int rb2 = (lane >= 16) ? 8 : 0;
#pragma unroll
    for (int r = 0; r < 8; ++r) {
      int q = q0 + rb2 + r;
      if (q < Q_) {
        float inv = 1.f / lrow[r];
        bf16_t* op = Ob + ((size_t)(b * Q_ + q)) * D_ + h * HD_;
        op[col]      = (bf16_t)(c0[r] * inv);
        op[16 + col] = (bf16_t)(c1[r] * inv);
      }
    }
  }
}

// -------------------------- deformable sampling ----------------------------
// one wave per (b,q,h); 32 lanes = HD channels (coalesced gathers).
__global__ __launch_bounds__(256) void deform_kernel(
    const float* __restrict__ refp, const bf16_t* __restrict__ value,
    const float* __restrict__ offb, const float* __restrict__ awb,
    bf16_t* __restrict__ outp) {
  const int lane = threadIdx.x & 31;
  const int wv   = threadIdx.x >> 5;
  int gw = blockIdx.x * 8 + wv;
  if (gw >= NQ_ * NH_) return;
  int h  = gw & 7;
  int nq = gw >> 3;
  int b  = nq / Q_;
  const float* rp = refp + (size_t)nq * 4;
  float cx = rp[0], cy = rp[1], rw = rp[2], rh = rp[3];
  const float* offq = offb + (size_t)nq * 256 + h * 32;   // [NLEV][NPTS][2]
  const float* awq  = awb  + (size_t)nq * 128 + h * 16;   // softmaxed
  const int Hs[4] = {80, 40, 20, 10};
  const int Ss[4] = {0, 6400, 8000, 8400};
  float acc = 0.f;
#pragma unroll
  for (int lvl = 0; lvl < 4; ++lvl) {
    const int Hl = Hs[lvl], Wl = Hs[lvl];
    const bf16_t* vb = value + ((size_t)(b * LMEM_ + Ss[lvl])) * 256 + h * 32 + lane;
#pragma unroll
    for (int pt = 0; pt < 4; ++pt) {
      float ox = offq[(lvl * 4 + pt) * 2 + 0];
      float oy = offq[(lvl * 4 + pt) * 2 + 1];
      float x = (cx + ox * 0.125f * rw) * (float)Wl - 0.5f;  // off/4*wh*0.5
      float y = (cy + oy * 0.125f * rh) * (float)Hl - 0.5f;
      float x0f = floorf(x), y0f = floorf(y);
      float fx = x - x0f, fy = y - y0f;
      int x0 = (int)x0f, y0 = (int)y0f;
      float aww = awq[lvl * 4 + pt];
      float sval = 0.f;
#pragma unroll
      for (int dy = 0; dy < 2; ++dy) {
#pragma unroll
        for (int dx = 0; dx < 2; ++dx) {
          int cxi = x0 + dx, cyi = y0 + dy;
          float w = (dx ? fx : 1.f - fx) * (dy ? fy : 1.f - fy);
          bool valid = (cxi >= 0) && (cxi < Wl) && (cyi >= 0) && (cyi < Hl);
          int ccx = cxi < 0 ? 0 : (cxi > Wl - 1 ? Wl - 1 : cxi);
          int ccy = cyi < 0 ? 0 : (cyi > Hl - 1 ? Hl - 1 : cyi);
          float v = (float)vb[(size_t)(ccy * Wl + ccx) * 256];
          sval += (valid ? w : 0.f) * v;
        }
      }
      acc += aww * sval;
    }
  }
  outp[(size_t)nq * 256 + h * 32 + lane] = (bf16_t)acc;
}

// ------------------------------- launcher ----------------------------------

extern "C" void kernel_launch(void* const* d_in, const int* in_sizes, int n_in,
                              void* d_out, int out_size, void* d_ws, size_t ws_size,
                              hipStream_t stream) {
  (void)in_sizes; (void)n_in; (void)out_size; (void)ws_size;
  const float* tgt  = (const float*)d_in[0];
  const float* refp = (const float*)d_in[1];
  const float* mem  = (const float*)d_in[2];
  const float* pos  = (const float*)d_in[5];
  const float* Wq   = (const float*)d_in[6];  const float* bq   = (const float*)d_in[7];
  const float* Wk   = (const float*)d_in[8];  const float* bk   = (const float*)d_in[9];
  const float* Wv   = (const float*)d_in[10]; const float* bv   = (const float*)d_in[11];
  const float* Wo   = (const float*)d_in[12]; const float* bo   = (const float*)d_in[13];
  const float* Woff = (const float*)d_in[14]; const float* boff = (const float*)d_in[15];
  const float* Waw  = (const float*)d_in[16]; const float* baw  = (const float*)d_in[17];
  const float* Wval = (const float*)d_in[18]; const float* bval = (const float*)d_in[19];
  const float* Wout = (const float*)d_in[20]; const float* bout = (const float*)d_in[21];
  const float* W1   = (const float*)d_in[22]; const float* b1   = (const float*)d_in[23];
  const float* W2   = (const float*)d_in[24]; const float* b2   = (const float*)d_in[25];
  const float* ln1g = (const float*)d_in[26]; const float* ln1b = (const float*)d_in[27];
  const float* ln2g = (const float*)d_in[28]; const float* ln2b = (const float*)d_in[29];
  const float* ln3g = (const float*)d_in[30]; const float* ln3b = (const float*)d_in[31];
  float* out = (float*)d_out;

  // ---- workspace carve (deterministic, no allocations) ----
  char* base = (char*)d_ws;
  size_t cur = 0;
  auto carve = [&](size_t bytes) -> void* {
    void* p = base + cur;
    cur += (bytes + 255) & ~(size_t)255;
    return p;
  };
  const size_t ND = (size_t)NQ_ * 256;
  bf16_t* qk_b    = (bf16_t*)carve(ND * 2);   // tgt+pos, bf16
  bf16_t* qbuf    = (bf16_t*)carve(ND * 2);   // pre-scaled Q
  bf16_t* kbuf    = (bf16_t*)carve(ND * 2);
  bf16_t* vbuf    = (bf16_t*)carve(ND * 2);
  bf16_t* obuf    = (bf16_t*)carve(ND * 2);
  float*  tgt1    = (float*)carve(ND * 4);
  bf16_t* qc_b    = (bf16_t*)carve(ND * 2);   // tgt1+pos, bf16
  bf16_t* sampled = (bf16_t*)carve(ND * 2);
  float*  tgt2    = (float*)carve(ND * 4);
  float*  offbuf  = (float*)carve(ND * 4);
  float*  awbuf   = (float*)carve((size_t)NQ_ * 128 * 4);
  bf16_t* ffn1    = (bf16_t*)carve((size_t)NQ_ * DFF_ * 2);
  bf16_t* valb    = (bf16_t*)carve((size_t)NM_ * 256 * 2);  // value tensor bf16
  float*  bqs     = (float*)carve(256 * 4);   // bq * 1/sqrt(HD)
  bf16_t* Wq_t    = (bf16_t*)carve(65536 * 2);  // weights bf16, [N,K]
  bf16_t* Wk_t    = (bf16_t*)carve(65536 * 2);
  bf16_t* Wv_t    = (bf16_t*)carve(65536 * 2);
  bf16_t* Wo_t    = (bf16_t*)carve(65536 * 2);
  bf16_t* Woff_t  = (bf16_t*)carve(65536 * 2);
  bf16_t* Waw_t   = (bf16_t*)carve(32768 * 2);
  bf16_t* Wval_t  = (bf16_t*)carve(65536 * 2);
  bf16_t* Wout_t  = (bf16_t*)carve(65536 * 2);
  bf16_t* W1_t    = (bf16_t*)carve(262144 * 2);
  bf16_t* W2_t    = (bf16_t*)carve(262144 * 2);

  const float qscale = 0.17677669529663687f;  // 1/sqrt(32)
  auto cvt = [&](const float* s, bf16_t* d, int K, int N, float sc) {
    int n = K * N;
    w_to_bf16_t<<<(n + 255) / 256, 256, 0, stream>>>(s, d, K, N, sc);
  };
  cvt(Wq, Wq_t, 256, 256, qscale);   cvt(Wk, Wk_t, 256, 256, 1.f);
  cvt(Wv, Wv_t, 256, 256, 1.f);      cvt(Wo, Wo_t, 256, 256, 1.f);
  cvt(Woff, Woff_t, 256, 256, 1.f);  cvt(Waw, Waw_t, 256, 128, 1.f);
  cvt(Wval, Wval_t, 256, 256, 1.f);  cvt(Wout, Wout_t, 256, 256, 1.f);
  cvt(W1, W1_t, 256, 1024, 1.f);     cvt(W2, W2_t, 1024, 256, 1.f);
  scale_kernel<<<1, 256, 0, stream>>>(bq, bqs, qscale, 256);

  // 1) self-attention
  add_bf16_kernel<<<(int)(ND / 256), 256, 0, stream>>>(tgt, pos, qk_b, (int)ND);
  gemm_tn<8, false, true, false, true><<<dim3(1, NQ_ / 64), 256, 0, stream>>>(
      nullptr, qk_b, Wq_t, bqs, nullptr, qbuf, NQ_, 256, 256,
      nullptr, nullptr, nullptr);
  gemm_tn<8, false, true, false, true><<<dim3(1, NQ_ / 64), 256, 0, stream>>>(
      nullptr, qk_b, Wk_t, bk, nullptr, kbuf, NQ_, 256, 256,
      nullptr, nullptr, nullptr);
  gemm_tn<8, false, true, false, false><<<dim3(1, NQ_ / 64), 256, 0, stream>>>(
      tgt, nullptr, Wv_t, bv, nullptr, vbuf, NQ_, 256, 256,
      nullptr, nullptr, nullptr);
  attn_flash<<<dim3(5, NH_, B_), 128, 0, stream>>>(qbuf, kbuf, vbuf, obuf);
  // Wo projection fused with residual+LN1 -> tgt1
  gemm_tn<8, false, false, true, true><<<dim3(1, NQ_ / 64), 256, 0, stream>>>(
      nullptr, obuf, Wo_t, bo, tgt1, nullptr, NQ_, 256, 256, tgt, ln1g, ln1b);

  // 2) deformable attention
  add_bf16_kernel<<<(int)(ND / 256), 256, 0, stream>>>(tgt1, pos, qc_b, (int)ND);
  gemm_tn<8, false, true, false, false><<<dim3(1, NM_ / 64), 256, 0, stream>>>(
      mem, nullptr, Wval_t, bval, nullptr, valb, NM_, 256, 256,
      nullptr, nullptr, nullptr);                               // one HBM pass
  gemm_tn<8, false, false, false, true><<<dim3(1, NQ_ / 64), 256, 0, stream>>>(
      nullptr, qc_b, Woff_t, boff, offbuf, nullptr, NQ_, 256, 256,
      nullptr, nullptr, nullptr);
  gemm_tn<4, false, false, false, true><<<dim3(1, NQ_ / 64), 256, 0, stream>>>(
      nullptr, qc_b, Waw_t, baw, awbuf, nullptr, NQ_, 128, 256,
      nullptr, nullptr, nullptr);
  softmax16_kernel<<<(NQ_ * NH_ + 255) / 256, 256, 0, stream>>>(awbuf, NQ_ * NH_);
  deform_kernel<<<(NQ_ * NH_) / 8, 256, 0, stream>>>(refp, valb, offbuf, awbuf, sampled);
  // Wout projection fused with residual+LN2 -> tgt2
  gemm_tn<8, false, false, true, true><<<dim3(1, NQ_ / 64), 256, 0, stream>>>(
      nullptr, sampled, Wout_t, bout, tgt2, nullptr, NQ_, 256, 256,
      tgt1, ln2g, ln2b);

  // 3) FFN (W1 -> bf16 with fused ReLU; W2 fused with residual+LN3 -> out)
  gemm_tn<8, true, true, false, false><<<dim3(4, NQ_ / 64), 256, 0, stream>>>(
      tgt2, nullptr, W1_t, b1, nullptr, ffn1, NQ_, DFF_, 256,
      nullptr, nullptr, nullptr);
  gemm_tn<8, false, false, true, true><<<dim3(1, NQ_ / 64), 256, 0, stream>>>(
      nullptr, ffn1, W2_t, b2, out, nullptr, NQ_, 256, 1024, tgt2, ln3g, ln3b);
}
